// MixtureofExperts_9534827397177
// MI455X (gfx1250) — matrix operand order repaired and checked by reading, not compiled, not run on hardware
//
#include <hip/hip_runtime.h>
#include <hip/hip_bf16.h>

// ---------------- types ----------------
typedef __attribute__((ext_vector_type(16))) __bf16 v16bf;
typedef __attribute__((ext_vector_type(8)))  float  v8f;

#define EMB 1024
#define HID 2048
#define NEXP 8
#define NTOK 16384

__device__ inline v8f vzero8() {
    v8f z;
#pragma unroll
    for (int i = 0; i < 8; ++i) z[i] = 0.f;
    return z;
}

// Build a 16x32 (16-bit) A fragment from two 16B chunks (callers use contiguous k = 16*half + s).
__device__ inline v16bf load_frag2(const __bf16* p0, const __bf16* p1) {
    union { int4 q[2]; v16bf v; } u;
    u.q[0] = *(const int4*)p0;
    u.q[1] = *(const int4*)p1;
    return u.v;
}

// ---------------- weight transpose + fp32->bf16 ----------------
// dst[c][r] = bf16(src[r][c]); src is R x C, one expert per blockIdx.z
__global__ void transpose_cvt_kernel(const float* __restrict__ src,
                                     __bf16* __restrict__ dst, int R, int C) {
    __shared__ float tile[32][33];
    const size_t esz = (size_t)R * C;
    src += (size_t)blockIdx.z * esz;
    dst += (size_t)blockIdx.z * esz;
    const int c0 = blockIdx.x * 32, r0 = blockIdx.y * 32;
#pragma unroll
    for (int i = threadIdx.y; i < 32; i += 8)
        tile[i][threadIdx.x] = src[(size_t)(r0 + i) * C + c0 + threadIdx.x];
    __syncthreads();
#pragma unroll
    for (int i = threadIdx.y; i < 32; i += 8)
        dst[(size_t)(c0 + i) * R + r0 + threadIdx.x] = (__bf16)tile[threadIdx.x][i];
}

// ---------------- router: logits, top-2 softmax, expert gather lists ----------------
// one wave (32 lanes) per token; also casts x -> bf16 workspace copy
__global__ void router_kernel(const float* __restrict__ x,
                              const float* __restrict__ rw,
                              __bf16* __restrict__ xb,
                              int* __restrict__ list,
                              float* __restrict__ gw,
                              int* __restrict__ cnt) {
    const int lane = threadIdx.x & 31;
    const int wave = threadIdx.x >> 5;
    const int t = blockIdx.x * 8 + wave;
    const float* xrow = x + (size_t)t * EMB;
    float acc[NEXP];
#pragma unroll
    for (int e = 0; e < NEXP; ++e) acc[e] = 0.f;

    for (int d = lane; d < EMB; d += 32) {
        float xv = xrow[d];
        xb[(size_t)t * EMB + d] = (__bf16)xv;
#pragma unroll
        for (int e = 0; e < NEXP; ++e) acc[e] += xv * rw[d * NEXP + e];
    }
#pragma unroll
    for (int e = 0; e < NEXP; ++e) {
        float v = acc[e];
#pragma unroll
        for (int off = 16; off > 0; off >>= 1) v += __shfl_xor(v, off, 32);
        acc[e] = v;
    }
    if (lane == 0) {
        int e1 = 0;
#pragma unroll
        for (int e = 1; e < NEXP; ++e) if (acc[e] > acc[e1]) e1 = e;
        int e2 = (e1 == 0) ? 1 : 0;
#pragma unroll
        for (int e = 0; e < NEXP; ++e) if (e != e1 && acc[e] > acc[e2]) e2 = e;
        const float v1 = acc[e1], v2 = acc[e2];
        const float q  = __expf(v2 - v1);
        const float p2 = q * __builtin_amdgcn_rcpf(1.f + q);
        const float p1 = 1.f - p2;
        int pos1 = atomicAdd(&cnt[e1], 1);
        list[e1 * NTOK + pos1] = t; gw[e1 * NTOK + pos1] = p1;
        int pos2 = atomicAdd(&cnt[e2], 1);
        list[e2 * NTOK + pos2] = t; gw[e2 * NTOK + pos2] = p2;
    }
}

// ---------------- fused expert kernel ----------------
// 64 gathered tokens per block; each wave owns ALL 64 M-rows (4 A fragments)
// and a private 32-column N strip, so each weight fragment is loaded from L2
// exactly once per block. SwiGLU row h[64x2048] staged in 256KB LDS.
__global__ __launch_bounds__(256, 1)
void moe_kernel(const __bf16* __restrict__ xb, const __bf16* __restrict__ wg,
                const __bf16* __restrict__ wu, const __bf16* __restrict__ wd,
                const int* __restrict__ list, const float* __restrict__ gw,
                const int* __restrict__ cnt, float* __restrict__ out) {
    const int e  = blockIdx.y;
    const int i0 = blockIdx.x * 64;
    const int c  = cnt[e];
    if (i0 >= c) return;

    __shared__ int   stok[64];
    __shared__ float swt[64];
    extern __shared__ __align__(16) char smem_raw[];
    __bf16* hsh = (__bf16*)smem_raw;            // [64][2048] bf16 = 256KB

    if (threadIdx.x < 64) {
        const int idx = i0 + threadIdx.x;
        if (idx < c) { stok[threadIdx.x] = list[e * NTOK + idx];
                       swt[threadIdx.x] = gw[e * NTOK + idx]; }
        else         { stok[threadIdx.x] = 0; swt[threadIdx.x] = 0.f; }
    }
    __syncthreads();

    const int lane = threadIdx.x & 31;
    const int wn   = threadIdx.x >> 5;   // wave owns 32-col strip
    const int lr   = lane & 15;          // col (B/C) or row-within-16 (A)
    const int lh   = lane >> 4;          // lane half -> K sub-run

    const __bf16* arow[4];
#pragma unroll
    for (int ms = 0; ms < 4; ++ms)
        arow[ms] = xb + (size_t)stok[ms * 16 + lr] * EMB;

    const __bf16* wge = wg + (size_t)e * HID * EMB;
    const __bf16* wue = wu + (size_t)e * HID * EMB;
    const __bf16* wde = wd + (size_t)e * EMB * HID;

    // ---- Phase 1: h = silu(x Wg) * (x Wu) -> LDS ----
    for (int nit = 0; nit < 8; ++nit) {
        const int hbase = nit * 256 + wn * 32;      // this wave's 32 columns
        v8f accg[4][2], accu[4][2];
#pragma unroll
        for (int ms = 0; ms < 4; ++ms)
#pragma unroll
            for (int ns = 0; ns < 2; ++ns) { accg[ms][ns] = vzero8(); accu[ms][ns] = vzero8(); }

        for (int k0 = 0; k0 < EMB; k0 += 32) {
            v16bf a[4];
#pragma unroll
            for (int ms = 0; ms < 4; ++ms)
                a[ms] = load_frag2(arow[ms] + k0 + lh * 16,
                                   arow[ms] + k0 + lh * 16 + 8);
#pragma unroll
            for (int ns = 0; ns < 2; ++ns) {
                const int hcol = hbase + ns * 16 + lr;
                const v16bf bg = *(const v16bf*)(wge + (size_t)hcol * EMB + k0 + lh * 16);
#pragma unroll
                for (int ms = 0; ms < 4; ++ms)
                    accg[ms][ns] = __builtin_amdgcn_wmma_f32_16x16x32_bf16(
                        false, a[ms], false, bg, (short)0, accg[ms][ns], false, false);
                const v16bf bu = *(const v16bf*)(wue + (size_t)hcol * EMB + k0 + lh * 16);
#pragma unroll
                for (int ms = 0; ms < 4; ++ms)
                    accu[ms][ns] = __builtin_amdgcn_wmma_f32_16x16x32_bf16(
                        false, a[ms], false, bu, (short)0, accu[ms][ns], false, false);
            }
        }
#pragma unroll
        for (int ns = 0; ns < 2; ++ns) {
            const int hcol = hbase + ns * 16 + lr;
#pragma unroll
            for (int ms = 0; ms < 4; ++ms)
#pragma unroll
                for (int v = 0; v < 8; ++v) {
                    const int m = ms * 16 + v + 8 * lh;
                    const float g = accg[ms][ns][v];
                    // silu(g) = g * sigmoid(g); fast rcp (1 ulp) — noise vs bf16 rounding
                    const float s = __builtin_amdgcn_rcpf(1.f + __expf(-g));
                    hsh[m * HID + hcol] = (__bf16)(g * s * accu[ms][ns][v]);
                }
        }
    }
    __syncthreads();

    // ---- Phase 2: out[tok] += (h Wd) * weight ----
    const __bf16* hrow[4];
#pragma unroll
    for (int ms = 0; ms < 4; ++ms)
        hrow[ms] = hsh + (size_t)(ms * 16 + lr) * HID;

    for (int nit = 0; nit < 4; ++nit) {
        const int nbase = nit * 256 + wn * 32;
        v8f acc[4][2];
#pragma unroll
        for (int ms = 0; ms < 4; ++ms)
#pragma unroll
            for (int ns = 0; ns < 2; ++ns) acc[ms][ns] = vzero8();

        for (int k0 = 0; k0 < HID; k0 += 32) {
            v16bf a[4];
#pragma unroll
            for (int ms = 0; ms < 4; ++ms)
                a[ms] = load_frag2(hrow[ms] + k0 + lh * 16,
                                   hrow[ms] + k0 + lh * 16 + 8);   // ds_load_b128
#pragma unroll
            for (int ns = 0; ns < 2; ++ns) {
                const int ncol = nbase + ns * 16 + lr;
                const v16bf b = *(const v16bf*)(wde + (size_t)ncol * HID + k0 + lh * 16);
#pragma unroll
                for (int ms = 0; ms < 4; ++ms)
                    acc[ms][ns] = __builtin_amdgcn_wmma_f32_16x16x32_bf16(
                        false, a[ms], false, b, (short)0, acc[ms][ns], false, false);
            }
        }
#pragma unroll
        for (int ns = 0; ns < 2; ++ns) {
            const int ncol = nbase + ns * 16 + lr;
#pragma unroll
            for (int ms = 0; ms < 4; ++ms)
#pragma unroll
                for (int v = 0; v < 8; ++v) {
                    const int m = ms * 16 + v + 8 * lh;
                    atomicAdd(&out[(size_t)stok[m] * EMB + ncol], acc[ms][ns][v] * swt[m]);
                }
        }
    }
}

// ---------------- launch ----------------
extern "C" void kernel_launch(void* const* d_in, const int* in_sizes, int n_in,
                              void* d_out, int out_size, void* d_ws, size_t ws_size,
                              hipStream_t stream) {
    (void)in_sizes; (void)n_in; (void)ws_size;
    const float* x        = (const float*)d_in[0];
    const float* router_w = (const float*)d_in[1];
    const float* w_gate   = (const float*)d_in[2];
    const float* w_up     = (const float*)d_in[3];
    const float* w_down   = (const float*)d_in[4];
    float* out = (float*)d_out;

    const size_t WSEG = (size_t)NEXP * EMB * HID * 2;   // bf16 weight tensor bytes
    const size_t XSEG = (size_t)NTOK * EMB * 2;
    char* ws = (char*)d_ws;
    __bf16* xb  = (__bf16*)(ws);
    __bf16* wgp = (__bf16*)(ws + XSEG);
    __bf16* wup = (__bf16*)(ws + XSEG + WSEG);
    __bf16* wdp = (__bf16*)(ws + XSEG + 2 * WSEG);
    int*    lst = (int*)  (ws + XSEG + 3 * WSEG);
    float*  gwp = (float*)(ws + XSEG + 3 * WSEG + (size_t)NEXP * NTOK * 4);
    int*    cnt = (int*)  (ws + XSEG + 3 * WSEG + (size_t)NEXP * NTOK * 8);

    hipMemsetAsync(cnt, 0, 64, stream);
    hipMemsetAsync(d_out, 0, (size_t)out_size * sizeof(float), stream);

    const dim3 tb(32, 8);
    transpose_cvt_kernel<<<dim3(HID / 32, EMB / 32, NEXP), tb, 0, stream>>>(w_gate, wgp, EMB, HID);
    transpose_cvt_kernel<<<dim3(HID / 32, EMB / 32, NEXP), tb, 0, stream>>>(w_up,   wup, EMB, HID);
    transpose_cvt_kernel<<<dim3(EMB / 32, HID / 32, NEXP), tb, 0, stream>>>(w_down, wdp, HID, EMB);

    router_kernel<<<NTOK / 8, 256, 0, stream>>>(x, router_w, xb, lst, gwp, cnt);

    const int dyn_lds = 64 * HID * 2;  // 256KB h-tile in CDNA5's 320KB LDS
    hipFuncSetAttribute((const void*)moe_kernel,
                        hipFuncAttributeMaxDynamicSharedMemorySize, dyn_lds);
    moe_kernel<<<dim3(NTOK / 64, NEXP), 256, dyn_lds, stream>>>(
        xb, wgp, wup, wdp, lst, gwp, cnt, out);
}
